// MultiHeadAttention_57724360458720
// MI455X (gfx1250) — matrix operand order repaired and checked by reading, not compiled, not run on hardware
//
#include <hip/hip_runtime.h>
#include <hip/hip_bf16.h>

// ---------------------------------------------------------------------------
// MI455X (gfx1250) multi-head attention: bf16 WMMA + async-to-LDS staging.
// B=16, N=1024, H=8, d=128, MODEL=1024.
// ---------------------------------------------------------------------------

typedef __attribute__((ext_vector_type(16))) __bf16 v16bf;
typedef __attribute__((ext_vector_type(8)))  __bf16 v8bf;
typedef __attribute__((ext_vector_type(8)))  float  v8f;

union BF16Frag {
    v16bf v;
    v8bf  h[2];
};

__device__ __forceinline__ v8bf load8(const __bf16* p) {
    return *reinterpret_cast<const v8bf*>(p);
}
__device__ __forceinline__ v8f zero8() {
    v8f z;
    #pragma unroll
    for (int i = 0; i < 8; ++i) z[i] = 0.0f;
    return z;
}
__device__ __forceinline__ v8f wmma_bf16(const BF16Frag& a, const BF16Frag& b, v8f c) {
    // D = A(16x32 bf16) * B(32x16 bf16) + C(16x16 f32)
    return __builtin_amdgcn_wmma_f32_16x16x32_bf16(
        false, a.v, false, b.v, (short)0, c, false, false);
}

// CDNA5 async copy: 16 bytes global -> LDS, tracked by ASYNCcnt.
__device__ __forceinline__ void async_copy16(unsigned lds_off, const __bf16* gsrc) {
    asm volatile("global_load_async_to_lds_b128 %0, %1, off"
                 :
                 : "v"(lds_off), "v"((unsigned long long)(const void*)gsrc)
                 : "memory");
}
__device__ __forceinline__ void wait_async_le(int n) {
    if (n == 0)      asm volatile("s_wait_asynccnt 0x0" ::: "memory");
    else if (n == 2) asm volatile("s_wait_asynccnt 0x2" ::: "memory");
    else             asm volatile("s_wait_asynccnt 0x4" ::: "memory");
}

// ---------------------------------------------------------------------------
// Conversions
// ---------------------------------------------------------------------------
__global__ __launch_bounds__(256) void f32_to_bf16_kernel(
    const float* __restrict__ src, __bf16* __restrict__ dst, int n) {
    int i = blockIdx.x * blockDim.x + threadIdx.x;
    int stride = gridDim.x * blockDim.x;
    for (; i < n; i += stride) dst[i] = (__bf16)src[i];
}

// W: f32 [1024 x 1024] row-major (K x N)  ->  Wt: bf16 [N x K] (K contiguous)
__global__ __launch_bounds__(256) void transpose1024_to_bf16_kernel(
    const float* __restrict__ W, __bf16* __restrict__ Wt) {
    int idx = blockIdx.x * blockDim.x + threadIdx.x;   // exactly 1M threads
    int kk = idx >> 10;
    int n  = idx & 1023;
    Wt[(size_t)n * 1024 + kk] = (__bf16)W[idx];
}

// ---------------------------------------------------------------------------
// Generic bf16 GEMM:  C[M x Ncols] = A[M x K] @ Bt[Ncols x K]^T (+ bias) * scale
// mode 0: bf16 row-major; 1: bf16 scaled (Q); 2: bf16 as vT[B][H][dv][n] (V);
// mode 3: f32 row-major (output projection).
// Block = 256 thr = 8 waves; block tile 128x128; wave tile 16x128.
// B tile double-buffered in LDS via global_load_async_to_lds_b128;
// A fragments double-buffered in registers; K-loop manually unrolled by 2
// so all buffer/fragment selections are compile-time constants.
// ---------------------------------------------------------------------------
__device__ __forceinline__ void gemm_chunk(const __bf16 (*__restrict__ btile)[32],
                                           const BF16Frag& a, v8f* acc,
                                           int hf, int r) {
    #pragma unroll
    for (int t = 0; t < 8; ++t) {
        BF16Frag b;
        const __bf16* pb = &btile[t * 16 + r][hf * 8];
        b.h[0] = load8(pb);
        b.h[1] = load8(pb + 16);
        acc[t] = wmma_bf16(a, b, acc[t]);
    }
}

__global__ __launch_bounds__(256) void gemm_bf16_kernel(
    const __bf16* __restrict__ A, const __bf16* __restrict__ Bt,
    const float* __restrict__ bias,
    __bf16* __restrict__ Cb, float* __restrict__ Cf, __bf16* __restrict__ CvT,
    int Ncols, int K, int mode, float scale) {
    __shared__ __attribute__((aligned(64))) __bf16 blds[2][128][32];

    const int tid  = threadIdx.x;
    const int lane = tid & 31;
    const int wv   = tid >> 5;
    const int hf   = (lane >> 4) & 1;
    const int r    = lane & 15;
    const int row0 = blockIdx.y * 128 + wv * 16;
    const int col0 = blockIdx.x * 128;

    // Cooperative B-tile slice for this thread (2 x 16B per chunk).
    const int scol = tid >> 1;
    const int skq  = (tid & 1) * 16;
    const __bf16* sgsrc = Bt + (size_t)(col0 + scol) * K + skq;
    const unsigned bdst0 = (unsigned)(uintptr_t)(&blds[0][scol][skq]);
    const unsigned bdst1 = (unsigned)(uintptr_t)(&blds[1][scol][skq]);
    const __bf16* arow = A + (size_t)(row0 + r) * K + hf * 8;

    v8f acc[8];
    #pragma unroll
    for (int t = 0; t < 8; ++t) acc[t] = zero8();

    // Prologue: chunk 0 -> buf0 / afr0.
    async_copy16(bdst0, sgsrc);
    async_copy16(bdst0 + 16, sgsrc + 8);
    BF16Frag afr0, afr1;
    afr0.h[0] = load8(arow);
    afr0.h[1] = load8(arow + 16);

    for (int kc = 0; kc < K; kc += 64) {
        // ---- even chunk: compute buf0/afr0, prefetch kc+32 -> buf1/afr1 ----
        afr1.h[0] = load8(arow + kc + 32);
        afr1.h[1] = load8(arow + kc + 48);
        async_copy16(bdst1, sgsrc + kc + 32);
        async_copy16(bdst1 + 16, sgsrc + kc + 40);
        wait_async_le(2);
        __syncthreads();
        gemm_chunk(blds[0], afr0, acc, hf, r);
        __syncthreads();

        // ---- odd chunk: compute buf1/afr1, prefetch kc+64 -> buf0/afr0 ----
        if (kc + 64 < K) {
            afr0.h[0] = load8(arow + kc + 64);
            afr0.h[1] = load8(arow + kc + 80);
            async_copy16(bdst0, sgsrc + kc + 64);
            async_copy16(bdst0 + 16, sgsrc + kc + 72);
            wait_async_le(2);
        } else {
            wait_async_le(0);
        }
        __syncthreads();
        gemm_chunk(blds[1], afr1, acc, hf, r);
        __syncthreads();
    }

    #pragma unroll
    for (int t = 0; t < 8; ++t) {
        #pragma unroll
        for (int i = 0; i < 8; ++i) {
            int row = row0 + hf * 8 + i;
            int col = col0 + t * 16 + r;
            float val = acc[t][i];
            if (bias) val += bias[col];
            val *= scale;
            if (mode == 3) {
                Cf[(size_t)row * Ncols + col] = val;
            } else if (mode == 2) {
                int bb = row >> 10, n = row & 1023;
                int hh = col >> 7,  dv = col & 127;
                CvT[((size_t)((bb * 8 + hh) * 128 + dv)) * 1024 + n] = (__bf16)val;
            } else {
                Cb[(size_t)row * Ncols + col] = (__bf16)val;
            }
        }
    }
}

// ---------------------------------------------------------------------------
// Fused flash attention with async-LDS staging of K and V tiles.
// Grid: (N/128, H, B); block 256 = 8 waves; each wave owns 16 query rows.
// Per 32-key chunk: K tile (32x128) and V tile (128x32) are staged into
// double-buffered LDS shared by all 8 waves.
// ---------------------------------------------------------------------------
__device__ __forceinline__ void attn_chunk(
    const __bf16 (*__restrict__ ktile)[128], const __bf16 (*__restrict__ vtile)[32],
    const BF16Frag* qf, v8f* o, float* mrow, float* lrow,
    __bf16 (*__restrict__ plwv)[32], int hf, int r) {
    // ---- scores: S(16x32) = Q(16x128) @ K^T(128x32), two 16-key tiles ----
    v8f s0 = zero8(), s1 = zero8();
    #pragma unroll
    for (int c = 0; c < 4; ++c) {
        BF16Frag kf0, kf1;
        const __bf16* p0 = &ktile[r][c * 32 + hf * 8];
        kf0.h[0] = load8(p0);
        kf0.h[1] = load8(p0 + 16);
        const __bf16* p1 = &ktile[16 + r][c * 32 + hf * 8];
        kf1.h[0] = load8(p1);
        kf1.h[1] = load8(p1 + 16);
        s0 = wmma_bf16(qf[c], kf0, s0);
        s1 = wmma_bf16(qf[c], kf1, s1);
    }

    // ---- online softmax (rows live in half-wave lane groups of 16) ----
    #pragma unroll
    for (int i = 0; i < 8; ++i) {
        float cm = fmaxf(s0[i], s1[i]);
        #pragma unroll
        for (int off = 8; off; off >>= 1)
            cm = fmaxf(cm, __shfl_xor(cm, off, 16));
        float nm    = fmaxf(mrow[i], cm);
        float alpha = __expf(mrow[i] - nm);
        mrow[i] = nm;
        float p0v = __expf(s0[i] - nm);
        float p1v = __expf(s1[i] - nm);
        s0[i] = p0v;
        s1[i] = p1v;
        float rs = p0v + p1v;
        #pragma unroll
        for (int off = 8; off; off >>= 1)
            rs += __shfl_xor(rs, off, 16);
        lrow[i] = lrow[i] * alpha + rs;
        #pragma unroll
        for (int j = 0; j < 8; ++j) o[j][i] *= alpha;
    }

    // ---- P: C-layout f32 -> row-major bf16 (per-wave LDS) -> A fragment ----
    #pragma unroll
    for (int i = 0; i < 8; ++i) {
        plwv[hf * 8 + i][r]      = (__bf16)s0[i];
        plwv[hf * 8 + i][16 + r] = (__bf16)s1[i];
    }
    asm volatile("s_wait_dscnt 0x0" ::: "memory");   // same-wave LDS RAW
    BF16Frag pf;
    pf.h[0] = load8(&plwv[r][hf * 8]);
    pf.h[1] = load8(&plwv[r][hf * 8 + 16]);

    // ---- O(16x128) += P(16x32) @ V(32x128), 8 d-subtiles ----
    #pragma unroll
    for (int j = 0; j < 8; ++j) {
        BF16Frag vf;
        const __bf16* p = &vtile[j * 16 + r][hf * 8];
        vf.h[0] = load8(p);
        vf.h[1] = load8(p + 16);
        o[j] = wmma_bf16(pf, vf, o[j]);
    }
}

__global__ __launch_bounds__(256) void attn_kernel(
    const __bf16* __restrict__ qb, const __bf16* __restrict__ kb,
    const __bf16* __restrict__ vT, __bf16* __restrict__ heads) {
    __shared__ __attribute__((aligned(64))) __bf16 kt[2][32][128];   // 16 KB
    __shared__ __attribute__((aligned(64))) __bf16 vt[2][128][32];   // 16 KB
    __shared__ __attribute__((aligned(64))) __bf16 plds[8][16][32];  //  8 KB

    const int tid  = threadIdx.x;
    const int lane = tid & 31;
    const int wv   = tid >> 5;
    const int hf   = (lane >> 4) & 1;
    const int r    = lane & 15;
    const int b    = blockIdx.z;
    const int h    = blockIdx.y;
    const int q0   = blockIdx.x * 128 + wv * 16;

    // Cooperative copy slices (per thread, per chunk: 2x16B K + 2x16B V).
    const int krow = tid >> 3, kseg = (tid & 7) * 16;
    const int vrow = tid >> 1, vseg = (tid & 1) * 16;
    const __bf16* ksrc = kb + (size_t)(b * 1024 + krow) * 1024 + h * 128 + kseg;
    const __bf16* vsrc = vT + (size_t)((b * 8 + h) * 128 + vrow) * 1024 + vseg;
    const unsigned kdst0 = (unsigned)(uintptr_t)(&kt[0][krow][kseg]);
    const unsigned kdst1 = (unsigned)(uintptr_t)(&kt[1][krow][kseg]);
    const unsigned vdst0 = (unsigned)(uintptr_t)(&vt[0][vrow][vseg]);
    const unsigned vdst1 = (unsigned)(uintptr_t)(&vt[1][vrow][vseg]);

    // Prologue: stage chunk 0 into buffer 0.  (K rows advance by 1024 elems/key)
    async_copy16(kdst0, ksrc);
    async_copy16(kdst0 + 16, ksrc + 8);
    async_copy16(vdst0, vsrc);
    async_copy16(vdst0 + 16, vsrc + 8);

    // Q fragments (Q pre-scaled by 1/sqrt(d) at projection time).
    BF16Frag qf[4];
    {
        const __bf16* qrow = qb + (size_t)(b * 1024 + q0 + r) * 1024 + h * 128;
        #pragma unroll
        for (int c = 0; c < 4; ++c) {
            const __bf16* p = qrow + c * 32 + hf * 8;
            qf[c].h[0] = load8(p);
            qf[c].h[1] = load8(p + 16);
        }
    }

    v8f o[8];
    #pragma unroll
    for (int j = 0; j < 8; ++j) o[j] = zero8();
    float mrow[8], lrow[8];
    #pragma unroll
    for (int i = 0; i < 8; ++i) { mrow[i] = -3.0e38f; lrow[i] = 0.0f; }

    for (int kc = 0; kc < 1024; kc += 64) {
        // ---- even chunk (buf0); prefetch keys kc+32 -> buf1 ----
        async_copy16(kdst1, ksrc + (size_t)(kc + 32) * 1024);
        async_copy16(kdst1 + 16, ksrc + (size_t)(kc + 32) * 1024 + 8);
        async_copy16(vdst1, vsrc + kc + 32);
        async_copy16(vdst1 + 16, vsrc + kc + 40);
        wait_async_le(4);
        __syncthreads();
        attn_chunk(kt[0], vt[0], qf, o, mrow, lrow, plds[wv], hf, r);
        __syncthreads();

        // ---- odd chunk (buf1); prefetch keys kc+64 -> buf0 ----
        if (kc + 64 < 1024) {
            async_copy16(kdst0, ksrc + (size_t)(kc + 64) * 1024);
            async_copy16(kdst0 + 16, ksrc + (size_t)(kc + 64) * 1024 + 8);
            async_copy16(vdst0, vsrc + kc + 64);
            async_copy16(vdst0 + 16, vsrc + kc + 72);
            wait_async_le(4);
        } else {
            wait_async_le(0);
        }
        __syncthreads();
        attn_chunk(kt[1], vt[1], qf, o, mrow, lrow, plds[wv], hf, r);
        __syncthreads();
    }

    // ---- normalize and store heads [B,N,H*d] as bf16 ----
    float rl[8];
    #pragma unroll
    for (int i = 0; i < 8; ++i) rl[i] = 1.0f / lrow[i];
    #pragma unroll
    for (int j = 0; j < 8; ++j) {
        #pragma unroll
        for (int i = 0; i < 8; ++i) {
            int row = q0 + hf * 8 + i;
            int col = h * 128 + j * 16 + r;
            heads[(size_t)(b * 1024 + row) * 1024 + col] = (__bf16)(o[j][i] * rl[i]);
        }
    }
}

// ---------------------------------------------------------------------------
// Launch
// ---------------------------------------------------------------------------
extern "C" void kernel_launch(void* const* d_in, const int* in_sizes, int n_in,
                              void* d_out, int out_size, void* d_ws, size_t ws_size,
                              hipStream_t stream) {
    (void)in_sizes; (void)n_in; (void)out_size; (void)ws_size;
    const float* x  = (const float*)d_in[0];
    const float* Wq = (const float*)d_in[1];
    const float* bq = (const float*)d_in[2];
    const float* Wk = (const float*)d_in[3];
    const float* bk = (const float*)d_in[4];
    const float* Wv = (const float*)d_in[5];
    const float* bv = (const float*)d_in[6];
    const float* Wo = (const float*)d_in[7];
    float* out = (float*)d_out;

    char* ws = (char*)d_ws;
    __bf16* xb    = (__bf16*)(ws);                 // 32 MiB (x in bf16; reused as heads)
    __bf16* wqt   = (__bf16*)(ws + 33554432);      //  2 MiB
    __bf16* wkt   = (__bf16*)(ws + 35651584);
    __bf16* wvt   = (__bf16*)(ws + 37748736);
    __bf16* wot   = (__bf16*)(ws + 39845888);
    __bf16* qbuf  = (__bf16*)(ws + 41943040);      // 32 MiB [B*N, H*d], pre-scaled
    __bf16* kbuf  = (__bf16*)(ws + 75497472);      // 32 MiB [B*N, H*d]
    __bf16* vTbuf = (__bf16*)(ws + 109051904);     // 32 MiB [B, H, d, N]
    __bf16* heads = xb;                            // alias: x no longer needed

    // 1) conversions
    f32_to_bf16_kernel<<<8192, 256, 0, stream>>>(x, xb, 16 * 1024 * 1024);
    transpose1024_to_bf16_kernel<<<4096, 256, 0, stream>>>(Wq, wqt);
    transpose1024_to_bf16_kernel<<<4096, 256, 0, stream>>>(Wk, wkt);
    transpose1024_to_bf16_kernel<<<4096, 256, 0, stream>>>(Wv, wvt);
    transpose1024_to_bf16_kernel<<<4096, 256, 0, stream>>>(Wo, wot);

    // 2) projections (M=16384, N=1024, K=1024), block tile 128x128
    dim3 ggrid(8, 128);   // N/128 x M/128
    const float norm = 0.08838834764831845f;   // 1/sqrt(128), folded into Q
    gemm_bf16_kernel<<<ggrid, 256, 0, stream>>>(xb, wqt, bq, qbuf, nullptr, nullptr,
                                                1024, 1024, 1, norm);
    gemm_bf16_kernel<<<ggrid, 256, 0, stream>>>(xb, wkt, bk, kbuf, nullptr, nullptr,
                                                1024, 1024, 0, 1.0f);
    gemm_bf16_kernel<<<ggrid, 256, 0, stream>>>(xb, wvt, bv, nullptr, nullptr, vTbuf,
                                                1024, 1024, 2, 1.0f);

    // 3) fused attention
    attn_kernel<<<dim3(8, 8, 16), 256, 0, stream>>>(qbuf, kbuf, vTbuf, heads);

    // 4) output projection -> f32 d_out
    gemm_bf16_kernel<<<ggrid, 256, 0, stream>>>(heads, wot, nullptr, nullptr, out, nullptr,
                                                1024, 1024, 3, 1.0f);
}
